// Decoder_52733608460677
// MI455X (gfx1250) — matrix operand order repaired and checked by reading, not compiled, not run on hardware
//
#include <hip/hip_runtime.h>
#include <hip/hip_bf16.h>

// Problem dimensions (fixed by the reference).
constexpr int Vv = 32000, Ee = 512, Hh = 1024, Bb = 128, Ss = 128, Tt = 15;

typedef __attribute__((ext_vector_type(16))) __bf16        v16bf;
typedef __attribute__((ext_vector_type(8)))  float         v8f;
typedef __attribute__((ext_vector_type(4)))  unsigned int  u32x4;
typedef __hip_bfloat16 bf16;

union TilePack { u32x4 u[2]; v16bf v; };

// ---------------------------------------------------------------------------
// Generic C[M,N] = A[M,K] * W[N,K]^T (+bias), bf16 inputs, f32 out.
// One wave computes a 16 x (16*NB) strip of C: a single A tile (16x32) is
// reused against NB B tiles per k-step (2 + 2*NB b128 loads per NB WMMAs).
// Per-lane addressing follows the CDNA5 ISA 16-bit A(16x32)/B(32x16) VGPR
// layouts: VGPRs 0-3 and 4-7 each hold 8 K-contiguous halfs.
// ---------------------------------------------------------------------------
template <int NB>
__global__ __launch_bounds__(128) void wmma_gemm_bt_kernel(
    const bf16* __restrict__ A,      // [M,K] row-major
    const bf16* __restrict__ W,      // [N,K] row-major (B = W^T)
    const float* __restrict__ bias,  // [N] or nullptr
    float* __restrict__ C,           // row m at C + m*ldc
    int M, int N, int K, long long ldc)
{
  const int lane = (int)(threadIdx.x & 31u);
  const int wave = (int)(threadIdx.x >> 5);
  const int nGroups = (N >> 4) / NB;
  const int totalGroups = nGroups * (M >> 4);
  const int grp = (int)blockIdx.x * 4 + wave;
  if (grp >= totalGroups) return;                // wave-uniform: EXEC stays full
  const int nT0 = (grp % nGroups) * NB;
  const int mT  = grp / nGroups;
  const int r16 = lane & 15;
  const int hi  = lane >> 4;

  const bf16* aRow = A + (long long)(mT * 16 + r16) * K;  // matrix row (M)
  const bf16* wRow[NB];
#pragma unroll
  for (int i = 0; i < NB; ++i)
    wRow[i] = W + (long long)((nT0 + i) * 16 + r16) * K;  // matrix col (N)

  v8f acc[NB];
#pragma unroll
  for (int i = 0; i < NB; ++i) acc[i] = (v8f){};

  for (int k0 = 0; k0 < K; k0 += 32) {
    TilePack a;
    // A 16x32 layout: lanes<16 hold K = k0+{0..7, 16..23}; lanes>=16 +8.
    a.u[0] = *(const u32x4*)(aRow + k0 + 8 * hi);
    a.u[1] = *(const u32x4*)(aRow + k0 + 16 + 8 * hi);
    TilePack b[NB];
#pragma unroll
    for (int i = 0; i < NB; ++i) {
      // B 32x16 layout: lanes<16 hold K = k0+{0..7, 16..23}; lanes>=16 +8.
      b[i].u[0] = *(const u32x4*)(wRow[i] + k0 + 8 * hi);
      b[i].u[1] = *(const u32x4*)(wRow[i] + k0 + 16 + 8 * hi);
    }
#pragma unroll
    for (int i = 0; i < NB; ++i)
      acc[i] = __builtin_amdgcn_wmma_f32_16x16x32_bf16(false, a.v, false, b[i].v,
                                                       (short)0, acc[i], false, false);
  }

  // C/D layout: VGPR r -> row r (lanes 0-15) / row 8+r (lanes 16-31).
#pragma unroll
  for (int i = 0; i < NB; ++i) {
    const int col = (nT0 + i) * 16 + r16;
    const float bv = bias ? bias[col] : 0.0f;
    float* cBase = C + (long long)(mT * 16 + 8 * hi) * ldc + col;
#pragma unroll
    for (int r = 0; r < 8; ++r)
      cBase[(long long)r * ldc] = acc[i][r] + bv;
  }
}

// ---------------------------------------------------------------------------
__global__ void f32_to_bf16_kernel(const float* __restrict__ in,
                                   bf16* __restrict__ out, long long n) {
  long long i = (long long)blockIdx.x * blockDim.x + threadIdx.x;
  if (i < n) out[i] = __float2bfloat16(in[i]);
}

__global__ void init_state_kernel(const float* __restrict__ ehs,
                                  float* __restrict__ h0, float* __restrict__ h1,
                                  bf16* __restrict__ h0b, bf16* __restrict__ h1b,
                                  int* __restrict__ tok) {
  int i = blockIdx.x * blockDim.x + threadIdx.x;
  if (i < Bb * Hh) {
    float a = ehs[i], b = ehs[Bb * Hh + i];
    h0[i] = a; h1[i] = b;
    h0b[i] = __float2bfloat16(a); h1b[i] = __float2bfloat16(b);
  }
  if (i < Bb) tok[i] = 0;  // SOS token
}

__global__ void embed_kernel(const float* __restrict__ emb,
                             const int* __restrict__ tok,
                             bf16* __restrict__ x) {
  int i = blockIdx.x * blockDim.x + threadIdx.x;
  if (i >= Bb * Ee) return;
  int b = i / Ee, e = i - b * Ee;
  x[i] = __float2bfloat16(emb[(long long)tok[b] * Ee + e]);
}

__global__ void gru_combine_kernel(const float* __restrict__ gi,
                                   const float* __restrict__ gh,
                                   const float* __restrict__ bih,
                                   const float* __restrict__ bhh,
                                   const float* __restrict__ hprev,
                                   float* __restrict__ hnew,
                                   bf16* __restrict__ hnewb) {
  int i = blockIdx.x * blockDim.x + threadIdx.x;
  if (i >= Bb * Hh) return;
  int b = i / Hh, j = i - b * Hh;
  const float* gib = gi + (long long)b * 3 * Hh;
  const float* ghb = gh + (long long)b * 3 * Hh;
  float ir = gib[j]          + bih[j],          hr = ghb[j]          + bhh[j];
  float iz = gib[Hh + j]     + bih[Hh + j],     hz = ghb[Hh + j]     + bhh[Hh + j];
  float in_ = gib[2*Hh + j]  + bih[2*Hh + j],   hn = ghb[2*Hh + j]   + bhh[2*Hh + j];
  float r = 1.0f / (1.0f + expf(-(ir + hr)));
  float z = 1.0f / (1.0f + expf(-(iz + hz)));
  float n = tanhf(in_ + r * hn);
  float h = (1.0f - z) * n + z * hprev[i];
  hnew[i] = h;
  hnewb[i] = __float2bfloat16(h);
}

// One block (128 threads == S) per batch row: scores, softmax, context,
// and emit cat = [h1, ctx] in bf16 for the W1 GEMM.
__global__ __launch_bounds__(128) void attention_kernel(
    const float* __restrict__ enc,   // [B,S,H]
    const float* __restrict__ h1,    // [B,H]
    bf16* __restrict__ cat)          // [B,2H]
{
  __shared__ float shh[Hh];
  __shared__ float shw[Ss];
  __shared__ float red[Ss];
  const int b = blockIdx.x, tid = threadIdx.x;

  for (int i = tid; i < Hh; i += Ss) {
    float v = h1[(long long)b * Hh + i];
    shh[i] = v;
    cat[(long long)b * 2 * Hh + i] = __float2bfloat16(v);
  }
  __syncthreads();

  const float* erow = enc + ((long long)b * Ss + tid) * Hh;
  float sc = 0.0f;
  for (int k = 0; k < Hh; ++k) sc += erow[k] * shh[k];

  red[tid] = sc; __syncthreads();
  for (int s = Ss / 2; s > 0; s >>= 1) {
    if (tid < s) red[tid] = fmaxf(red[tid], red[tid + s]);
    __syncthreads();
  }
  float mx = red[0]; __syncthreads();

  float e = expf(sc - mx);
  shw[tid] = e;
  red[tid] = e; __syncthreads();
  for (int s = Ss / 2; s > 0; s >>= 1) {
    if (tid < s) red[tid] += red[tid + s];
    __syncthreads();
  }
  float inv = 1.0f / red[0];
  __syncthreads();

  for (int h = tid; h < Hh; h += Ss) {
    const float* ecol = enc + (long long)b * Ss * Hh + h;
    float acc = 0.0f;
    for (int s = 0; s < Ss; ++s) acc += shw[s] * ecol[(long long)s * Hh];
    cat[(long long)b * 2 * Hh + Hh + h] = __float2bfloat16(acc * inv);
  }
}

__global__ void tanh_bf_kernel(const float* __restrict__ in, bf16* __restrict__ out) {
  int i = blockIdx.x * blockDim.x + threadIdx.x;
  if (i < Bb * Hh) out[i] = __float2bfloat16(tanhf(in[i]));
}

// In-place log-softmax over V per row + argmax -> next token.
__global__ __launch_bounds__(256) void logsm_argmax_kernel(
    float* __restrict__ base,       // d_out + t*V; row stride = T*V
    int* __restrict__ tok)
{
  __shared__ float smax[256]; __shared__ int sidx[256]; __shared__ float ssum[256];
  const int b = blockIdx.x, tid = threadIdx.x;
  float* row = base + (long long)b * Tt * Vv;

  float mx = -3.0e38f; int mi = 0;
  for (int i = tid; i < Vv; i += 256) {
    float v = row[i];
    if (v > mx) { mx = v; mi = i; }
  }
  smax[tid] = mx; sidx[tid] = mi; __syncthreads();
  for (int s = 128; s > 0; s >>= 1) {
    if (tid < s) {
      if (smax[tid + s] > smax[tid] ||
          (smax[tid + s] == smax[tid] && sidx[tid + s] < sidx[tid])) {
        smax[tid] = smax[tid + s]; sidx[tid] = sidx[tid + s];
      }
    }
    __syncthreads();
  }
  float gmax = smax[0];

  float sum = 0.0f;
  for (int i = tid; i < Vv; i += 256) sum += expf(row[i] - gmax);
  ssum[tid] = sum; __syncthreads();
  for (int s = 128; s > 0; s >>= 1) {
    if (tid < s) ssum[tid] += ssum[tid + s];
    __syncthreads();
  }
  float lse = gmax + logf(ssum[0]);

  for (int i = tid; i < Vv; i += 256) row[i] -= lse;
  if (tid == 0) tok[b] = sidx[0];
}

__global__ void final_state_kernel(const float* __restrict__ h0,
                                   const float* __restrict__ h1,
                                   float* __restrict__ out) {
  int i = blockIdx.x * blockDim.x + threadIdx.x;
  if (i >= 2 * Bb * Hh) return;
  out[(long long)Bb * Tt * Vv + i] = (i < Bb * Hh) ? h0[i] : h1[i - Bb * Hh];
}

// ---------------------------------------------------------------------------
extern "C" void kernel_launch(void* const* d_in, const int* in_sizes, int n_in,
                              void* d_out, int out_size, void* d_ws, size_t ws_size,
                              hipStream_t stream) {
  (void)in_sizes; (void)n_in; (void)out_size; (void)ws_size;
  const float* ehs  = (const float*)d_in[1];
  const float* enc  = (const float*)d_in[2];
  const float* emb  = (const float*)d_in[3];
  const float* Wih0 = (const float*)d_in[4];
  const float* Whh0 = (const float*)d_in[5];
  const float* bih0 = (const float*)d_in[6];
  const float* bhh0 = (const float*)d_in[7];
  const float* Wih1 = (const float*)d_in[8];
  const float* Whh1 = (const float*)d_in[9];
  const float* bih1 = (const float*)d_in[10];
  const float* bhh1 = (const float*)d_in[11];
  const float* W1   = (const float*)d_in[12];
  const float* W2   = (const float*)d_in[13];
  const float* b2   = (const float*)d_in[14];
  float* out = (float*)d_out;

  size_t off = 0;
  auto take = [&](size_t bytes) -> void* {
    void* p = (char*)d_ws + off;
    off += (bytes + 255) & ~(size_t)255;
    return p;
  };
  bf16* Wih0b = (bf16*)take((size_t)3 * Hh * Ee * 2);
  bf16* Whh0b = (bf16*)take((size_t)3 * Hh * Hh * 2);
  bf16* Wih1b = (bf16*)take((size_t)3 * Hh * Hh * 2);
  bf16* Whh1b = (bf16*)take((size_t)3 * Hh * Hh * 2);
  bf16* W1b   = (bf16*)take((size_t)Hh * 2 * Hh * 2);
  bf16* W2b   = (bf16*)take((size_t)Vv * Hh * 2);        // 65.5 MB, L2-resident
  int*  tok   = (int*) take((size_t)Bb * 4);
  bf16* xb    = (bf16*)take((size_t)Bb * Ee * 2);
  float* h0   = (float*)take((size_t)Bb * Hh * 4);
  float* h1   = (float*)take((size_t)Bb * Hh * 4);
  bf16* h0b   = (bf16*)take((size_t)Bb * Hh * 2);
  bf16* h1b   = (bf16*)take((size_t)Bb * Hh * 2);
  float* gi   = (float*)take((size_t)Bb * 3 * Hh * 4);
  float* gh   = (float*)take((size_t)Bb * 3 * Hh * 4);
  bf16* catb  = (bf16*)take((size_t)Bb * 2 * Hh * 2);
  float* of32 = (float*)take((size_t)Bb * Hh * 4);
  bf16* ob    = (bf16*)take((size_t)Bb * Hh * 2);

  auto cvt = [&](const float* src, bf16* dst, long long n) {
    f32_to_bf16_kernel<<<dim3((unsigned)((n + 255) / 256)), dim3(256), 0, stream>>>(src, dst, n);
  };
  auto gemm1 = [&](const bf16* A, const bf16* Wm, const float* bias, float* C,
                   int M, int N, int K, long long ldc) {
    int groups = (N >> 4) * (M >> 4);
    wmma_gemm_bt_kernel<1><<<dim3((groups + 3) / 4), dim3(128), 0, stream>>>(
        A, Wm, bias, C, M, N, K, ldc);
  };
  auto gemm2 = [&](const bf16* A, const bf16* Wm, const float* bias, float* C,
                   int M, int N, int K, long long ldc) {
    int groups = ((N >> 4) / 2) * (M >> 4);
    wmma_gemm_bt_kernel<2><<<dim3((groups + 3) / 4), dim3(128), 0, stream>>>(
        A, Wm, bias, C, M, N, K, ldc);
  };
  auto gemm4 = [&](const bf16* A, const bf16* Wm, const float* bias, float* C,
                   int M, int N, int K, long long ldc) {
    int groups = ((N >> 4) / 4) * (M >> 4);
    wmma_gemm_bt_kernel<4><<<dim3((groups + 3) / 4), dim3(128), 0, stream>>>(
        A, Wm, bias, C, M, N, K, ldc);
  };

  // Weight precision pre-pass (every call; deterministic).
  cvt(Wih0, Wih0b, (long long)3 * Hh * Ee);
  cvt(Whh0, Whh0b, (long long)3 * Hh * Hh);
  cvt(Wih1, Wih1b, (long long)3 * Hh * Hh);
  cvt(Whh1, Whh1b, (long long)3 * Hh * Hh);
  cvt(W1,   W1b,   (long long)Hh * 2 * Hh);
  cvt(W2,   W2b,   (long long)Vv * Hh);

  init_state_kernel<<<dim3((Bb * Hh + 255) / 256), dim3(256), 0, stream>>>(
      ehs, h0, h1, h0b, h1b, tok);

  for (int t = 0; t < Tt; ++t) {
    embed_kernel<<<dim3((Bb * Ee + 255) / 256), dim3(256), 0, stream>>>(emb, tok, xb);

    // GRU layer 0
    gemm2(xb,  Wih0b, nullptr, gi, Bb, 3 * Hh, Ee, 3 * Hh);
    gemm2(h0b, Whh0b, nullptr, gh, Bb, 3 * Hh, Hh, 3 * Hh);
    gru_combine_kernel<<<dim3((Bb * Hh + 255) / 256), dim3(256), 0, stream>>>(
        gi, gh, bih0, bhh0, h0, h0, h0b);

    // GRU layer 1 (input = new h0; hidden = previous h1)
    gemm2(h0b, Wih1b, nullptr, gi, Bb, 3 * Hh, Hh, 3 * Hh);
    gemm2(h1b, Whh1b, nullptr, gh, Bb, 3 * Hh, Hh, 3 * Hh);
    gru_combine_kernel<<<dim3((Bb * Hh + 255) / 256), dim3(256), 0, stream>>>(
        gi, gh, bih1, bhh1, h1, h1, h1b);

    // Attention + concat
    attention_kernel<<<dim3(Bb), dim3(Ss), 0, stream>>>(enc, h1, catb);

    // o = tanh(cat @ W1^T)  (small; keep unblocked for occupancy)
    gemm1(catb, W1b, nullptr, of32, Bb, Hh, 2 * Hh, Hh);
    tanh_bf_kernel<<<dim3((Bb * Hh + 255) / 256), dim3(256), 0, stream>>>(of32, ob);

    // Vocab GEMM straight into d_out[:, t, :], then in-place log-softmax+argmax.
    gemm4(ob, W2b, b2, out + (long long)t * Vv, Bb, Vv, Hh, (long long)Tt * Vv);
    logsm_argmax_kernel<<<dim3(Bb), dim3(256), 0, stream>>>(out + (long long)t * Vv, tok);
  }

  final_state_kernel<<<dim3((2 * Bb * Hh + 255) / 256), dim3(256), 0, stream>>>(h0, h1, out);
}
